// Generator_61478161875296
// MI455X (gfx1250) — compile-verified, matrix operand order checked by reading, not run on hardware
//
#include <hip/hip_runtime.h>
#include <hip/hip_bf16.h>
#include <math.h>
#include <stdint.h>

#define B_  2
#define S_  1024
#define D_  512
#define H_  8
#define DH_ 64
#define L_  8
#define F_  2048
#define V_  32000
#define C_  4

typedef __attribute__((ext_vector_type(16))) __bf16        v16bf;
typedef __attribute__((ext_vector_type(8)))  float         v8f;
typedef __attribute__((ext_vector_type(8)))  unsigned int  v8u;
typedef __attribute__((ext_vector_type(4)))  unsigned int  u32x4;
typedef __attribute__((ext_vector_type(4)))  int           i32x4;
typedef __attribute__((ext_vector_type(8)))  int           i32x8;
typedef __attribute__((ext_vector_type(4)))  float         f32x4;

#if __has_builtin(__builtin_amdgcn_tensor_load_to_lds)
#define USE_TDM 1
#else
#define USE_TDM 0
#endif

// ---------------- helpers ----------------

__device__ __forceinline__ unsigned short f2bf_bits(float f) {
    union { float f; unsigned int u; } x; x.f = f;
    unsigned int r = x.u + 0x7FFFu + ((x.u >> 16) & 1u);   // round-to-nearest-even
    return (unsigned short)(r >> 16);
}

__device__ __forceinline__ unsigned int pack_bf2(float lo, float hi) {
    return (unsigned int)f2bf_bits(lo) | ((unsigned int)f2bf_bits(hi) << 16);
}

__device__ __forceinline__ float gelu_tanh(float x) {
    float x3 = x * x * x;
    return 0.5f * x * (1.0f + tanhf(0.7978845608028654f * (x + 0.044715f * x3)));
}

__device__ __forceinline__ float elu1(float x) {
    return x > 0.0f ? x + 1.0f : expf(x);
}

__device__ __forceinline__ float u01hash(unsigned int row, unsigned int col) {
    unsigned int x = row * 0x9E3779B1u ^ col * 0x85EBCA77u ^ 0x6C62272Eu;
    x ^= x >> 16; x *= 0x7FEB352Du;
    x ^= x >> 15; x *= 0x846CA68Bu;
    x ^= x >> 16;
    return (float)(x >> 8) * (1.0f / 16777216.0f) + 1e-7f;  // (0,1)
}

__device__ __forceinline__ float gumbel_noise(unsigned int row, unsigned int col) {
    return -logf(-logf(u01hash(row, col)));
}

// ---- TDM: 2-D tile load global(fp32) -> LDS, tracked by TENSORcnt ----
// tile_w elements per row, tile_h rows, row stride = stride_elems (fp32 units).
// tensor_dim set equal to tile dims (tile never goes OOB at our call sites).

__device__ __forceinline__ void tdm_load_2d(const float* gsrc, unsigned int lds_byte_off,
                                            unsigned int tile_w, unsigned int tile_h,
                                            unsigned int stride_elems)
{
#if USE_TDM
    unsigned long long ga = (unsigned long long)(uintptr_t)gsrc;
    u32x4 g0;
    g0[0] = 1u;                                            // count=1, user mode
    g0[1] = lds_byte_off;                                  // D#.lds_addr
    g0[2] = (unsigned int)(ga & 0xFFFFFFFFull);            // global_addr[31:0]
    g0[3] = (unsigned int)((ga >> 32) & 0x01FFFFFFull)     // global_addr[56:32]
          | (2u << 30);                                    // type = 2 ("image")
    i32x8 g1;
    g1[0] = (int)(2u << 16);                               // data_size = 4 bytes
    g1[1] = (int)((tile_w & 0xFFFFu) << 16);               // tensor_dim0[15:0]  @bits 79:48
    g1[2] = (int)(((tile_w >> 16) & 0xFFFFu)
          | ((tile_h & 0xFFFFu) << 16));                   // tensor_dim0[31:16] | tensor_dim1[15:0]
    g1[3] = (int)(((tile_h >> 16) & 0xFFFFu)
          | ((tile_w & 0xFFFFu) << 16));                   // tensor_dim1[31:16] | tile_dim0
    g1[4] = (int)(tile_h & 0xFFFFu);                       // tile_dim1 (tile_dim2 = 0)
    g1[5] = (int)stride_elems;                             // tensor_dim0_stride[31:0]
    g1[6] = 0;                                             // stride hi / dim1_stride lo
    g1[7] = 0;
    i32x4 z4 = {0, 0, 0, 0};
#if __clang_major__ >= 23
    i32x8 z8 = {0, 0, 0, 0, 0, 0, 0, 0};
    __builtin_amdgcn_tensor_load_to_lds(g0, g1, z4, z4, z8, 0);
#else
    __builtin_amdgcn_tensor_load_to_lds(g0, g1, z4, z4, 0);
#endif
#else
    (void)gsrc; (void)lds_byte_off; (void)tile_w; (void)tile_h; (void)stride_elems;
#endif
}

__device__ __forceinline__ void tdm_wait0() {
#if USE_TDM
#if __has_builtin(__builtin_amdgcn_s_wait_tensorcnt)
    __builtin_amdgcn_s_wait_tensorcnt(0);
#endif
#endif
}

// ---------------- WMMA GEMM:  C = epilogue(A[MxK] * B[KxN] + bias) ----------------
// EPI: 0 = bias, 1 = bias+gelu, 2 = bias+residual
// M % 64 == 0, N % 64 == 0, K % 32 == 0 (true for all call sites here)
// Block: 256 threads (8 waves), 64x64 macro tile, K stepped by 32.
// TDM double-buffers fp32 tiles into LDS; a convert pass packs bf16 operand tiles.

template <int EPI>
__global__ __launch_bounds__(256) void gemm_wmma(
    const float* __restrict__ A, const float* __restrict__ Bm,
    const float* __restrict__ bias, const float* __restrict__ res,
    float* __restrict__ C, int M, int N, int K)
{
    __shared__ unsigned short As[64][40];   // bf16 bits, [m][k]
    __shared__ unsigned short Bs[64][40];   // bf16 bits, [n][k]  (B transposed)
#if USE_TDM
    __shared__ float Afp[2][64 * 32];       // fp32 tile [m][k], packed
    __shared__ float Bfp[2][32 * 64];       // fp32 tile [k][n], packed
#endif

    const int tid  = threadIdx.x;
    const int lane = tid & 31;
    const int wave = tid >> 5;
    const int m0   = blockIdx.y * 64;
    const int n0   = blockIdx.x * 64;
    const int tr   = wave & 3;      // wave tile-row (m offset 16*tr)
    const int ch   = wave >> 2;     // wave col-half (n offset 32*ch)
    const int hh   = lane >> 4;
    const int l16  = lane & 15;

    // staging decomposition (per thread)
    const int ai = tid >> 2;            // A row 0..63
    const int ak = (tid & 3) * 8;       // A k-segment start
    const int bk = (tid >> 4) * 2;      // B k-pair 0,2,..,30
    const int bn = (tid & 15) * 4;      // B n-segment start

    v8f c0 = {};
    v8f c1 = {};
    const int nk = K / 32;

#if USE_TDM
    if (wave == 0) {
        tdm_load_2d(A + (size_t)m0 * K, (unsigned int)(uintptr_t)&Afp[0][0], 32, 64, (unsigned int)K);
        tdm_load_2d(Bm + (size_t)n0,    (unsigned int)(uintptr_t)&Bfp[0][0], 64, 32, (unsigned int)N);
        tdm_wait0();
    }
    __syncthreads();
#endif

    for (int kt = 0; kt < nk; ++kt) {
        const int k0 = kt * 32;
#if USE_TDM
        const int cur = kt & 1, nxt = cur ^ 1;
        if (wave == 0 && kt + 1 < nk) {   // kick off next tiles while we convert+compute
            tdm_load_2d(A + (size_t)m0 * K + (k0 + 32),
                        (unsigned int)(uintptr_t)&Afp[nxt][0], 32, 64, (unsigned int)K);
            tdm_load_2d(Bm + (size_t)(k0 + 32) * N + n0,
                        (unsigned int)(uintptr_t)&Bfp[nxt][0], 64, 32, (unsigned int)N);
        }
        // ---- convert fp32 LDS tiles -> bf16 operand tiles ----
        {
            f32x4 a0 = *(const f32x4*)&Afp[cur][ai * 32 + ak];
            f32x4 a1 = *(const f32x4*)&Afp[cur][ai * 32 + ak + 4];
            u32x4 w;
            w[0] = pack_bf2(a0.x, a0.y);
            w[1] = pack_bf2(a0.z, a0.w);
            w[2] = pack_bf2(a1.x, a1.y);
            w[3] = pack_bf2(a1.z, a1.w);
            *(u32x4*)&As[ai][ak] = w;    // 16B aligned: row stride 80B, ak*2 in {0,16,32,48}

            f32x4 r0 = *(const f32x4*)&Bfp[cur][bk * 64 + bn];
            f32x4 r1 = *(const f32x4*)&Bfp[cur][(bk + 1) * 64 + bn];
            *(unsigned int*)&Bs[bn + 0][bk] = pack_bf2(r0.x, r1.x);
            *(unsigned int*)&Bs[bn + 1][bk] = pack_bf2(r0.y, r1.y);
            *(unsigned int*)&Bs[bn + 2][bk] = pack_bf2(r0.z, r1.z);
            *(unsigned int*)&Bs[bn + 3][bk] = pack_bf2(r0.w, r1.w);
        }
#else
        // ---- fallback: vectorized global staging ----
        {
            f32x4 a0 = *(const f32x4*)&A[(size_t)(m0 + ai) * K + k0 + ak];
            f32x4 a1 = *(const f32x4*)&A[(size_t)(m0 + ai) * K + k0 + ak + 4];
            u32x4 w;
            w[0] = pack_bf2(a0.x, a0.y);
            w[1] = pack_bf2(a0.z, a0.w);
            w[2] = pack_bf2(a1.x, a1.y);
            w[3] = pack_bf2(a1.z, a1.w);
            *(u32x4*)&As[ai][ak] = w;

            f32x4 r0 = *(const f32x4*)&Bm[(size_t)(k0 + bk) * N + n0 + bn];
            f32x4 r1 = *(const f32x4*)&Bm[(size_t)(k0 + bk + 1) * N + n0 + bn];
            *(unsigned int*)&Bs[bn + 0][bk] = pack_bf2(r0.x, r1.x);
            *(unsigned int*)&Bs[bn + 1][bk] = pack_bf2(r0.y, r1.y);
            *(unsigned int*)&Bs[bn + 2][bk] = pack_bf2(r0.z, r1.z);
            *(unsigned int*)&Bs[bn + 3][bk] = pack_bf2(r0.w, r1.w);
        }
        if (k0 + 32 < K) {
            __builtin_prefetch(&A[(size_t)(m0 + ai) * K + k0 + 32 + ak], 0, 1);
            __builtin_prefetch(&Bm[(size_t)(k0 + 32 + bk) * N + n0 + bn], 0, 1);
        }
#endif
        __syncthreads();   // bf16 tiles ready

        // ---- fragments: 16-bit layout, k = 2j + 8h (+8 for j>=4) ----
        v8u au, bu0, bu1;
        const int am  = tr * 16 + l16;
        const int bnf = ch * 32 + l16;
        #pragma unroll
        for (int j = 0; j < 8; ++j) {
            int kk = 2 * j + 8 * hh + (j >= 4 ? 8 : 0);
            au[j]  = *(const unsigned int*)&As[am][kk];
            bu0[j] = *(const unsigned int*)&Bs[bnf][kk];
            bu1[j] = *(const unsigned int*)&Bs[bnf + 16][kk];
        }
        v16bf af  = __builtin_bit_cast(v16bf, au);
        v16bf bf0 = __builtin_bit_cast(v16bf, bu0);
        v16bf bf1 = __builtin_bit_cast(v16bf, bu1);

        c0 = __builtin_amdgcn_wmma_f32_16x16x32_bf16(false, af, false, bf0,
                                                     (short)0, c0, false, false);
        c1 = __builtin_amdgcn_wmma_f32_16x16x32_bf16(false, af, false, bf1,
                                                     (short)0, c1, false, false);

#if USE_TDM
        if (wave == 0 && kt + 1 < nk) tdm_wait0();   // next fp32 tiles landed
#endif
        __syncthreads();   // everyone done reading As/Bs (and TDM data published)
    }

    // ---- epilogue: D layout lane=(n%16, half), vgpr r -> m = r + 8*half ----
    #pragma unroll
    for (int r = 0; r < 8; ++r) {
        int m   = m0 + tr * 16 + r + 8 * hh;
        int n_a = n0 + ch * 32 + l16;
        int n_b = n_a + 16;
        float va = c0[r] + bias[n_a];
        float vb = c1[r] + bias[n_b];
        if (EPI == 1) { va = gelu_tanh(va); vb = gelu_tanh(vb); }
        if (EPI == 2) {
            va += res[(size_t)m * N + n_a];
            vb += res[(size_t)m * N + n_b];
        }
        C[(size_t)m * N + n_a] = va;
        C[(size_t)m * N + n_b] = vb;
    }
}

// ---------------- embedding + positional ----------------

__global__ __launch_bounds__(256) void embed_kernel(
    const int* __restrict__ inp, const float* __restrict__ emb,
    const float* __restrict__ pos, float* __restrict__ x)
{
    int row = blockIdx.x;           // B*S
    int t   = threadIdx.x;          // 256, 2 elems each
    int tok = inp[row];
    int s   = row % S_;
    x[(size_t)row * D_ + t]        = emb[(size_t)tok * D_ + t]        + pos[(size_t)s * D_ + t];
    x[(size_t)row * D_ + t + 256]  = emb[(size_t)tok * D_ + t + 256]  + pos[(size_t)s * D_ + t + 256];
}

// ---------------- RoPE + elu+1 feature map (q), masked (k) ----------------

__global__ __launch_bounds__(256) void rope_elu_kernel(
    float* __restrict__ q, float* __restrict__ k, const float* __restrict__ mask)
{
    int idx = blockIdx.x * blockDim.x + threadIdx.x;   // B*S*H*32 pairs
    if (idx >= B_ * S_ * H_ * 32) return;
    int dr  = idx & 31;
    int rem = idx >> 5;
    int h   = rem % H_; rem /= H_;
    int s   = rem % S_;
    int b   = rem / S_;

    float inv = expf(-((float)(2 * dr) / (float)DH_) * logf(10000.0f));
    float ang = (float)s * inv;
    float cs = cosf(ang), sn = sinf(ang);

    size_t base = (((size_t)b * S_ + s) * H_ + h) * DH_;
    size_t i0 = base + dr, i1 = base + dr + 32;

    float q0 = q[i0], q1 = q[i1];
    float k0 = k[i0], k1 = k[i1];
    float qo0 = q0 * cs - q1 * sn;   // rot_half: d<32 -> -t2
    float qo1 = q1 * cs + q0 * sn;   // d>=32 -> +t1
    float ko0 = k0 * cs - k1 * sn;
    float ko1 = k1 * cs + k0 * sn;
    float m = mask[(size_t)b * S_ + s];

    q[i0] = elu1(qo0);      q[i1] = elu1(qo1);
    k[i0] = elu1(ko0) * m;  k[i1] = elu1(ko1) * m;
}

// ---------------- causal linear-attention scan ----------------
// one block per (b,h), 64 threads; thread e owns column e of the 64x64 kv state

__global__ __launch_bounds__(64) void la_scan_kernel(
    const float* __restrict__ q, const float* __restrict__ k,
    const float* __restrict__ v, float* __restrict__ a)
{
    int bh = blockIdx.x;
    int b  = bh / H_;
    int h  = bh % H_;
    int e  = threadIdx.x;  // 0..63

    float kv[DH_];
    float z[DH_];
    #pragma unroll
    for (int d = 0; d < DH_; ++d) { kv[d] = 0.0f; z[d] = 0.0f; }

    __shared__ float qs[DH_], ks[DH_], vs[DH_];

    for (int s = 0; s < S_; ++s) {
        size_t base = (((size_t)b * S_ + s) * H_ + h) * DH_;
        qs[e] = q[base + e];
        ks[e] = k[base + e];
        vs[e] = v[base + e];
        __syncthreads();
        float ve = vs[e];
        float num = 0.0f, den = 0.0f;
        #pragma unroll
        for (int d = 0; d < DH_; ++d) {
            kv[d] += ks[d] * ve;
            z[d]  += ks[d];
            num   += qs[d] * kv[d];
            den   += qs[d] * z[d];
        }
        a[base + e] = num / (den + 1e-6f);
        __syncthreads();
    }
}

// ---------------- conditional layer-norm ----------------

__global__ __launch_bounds__(256) void cond_ln_kernel(
    const float* __restrict__ x, const float* __restrict__ g,
    const float* __restrict__ bt, const int* __restrict__ cond,
    float* __restrict__ out)
{
    __shared__ float ssum[256], ssq[256];
    int row = blockIdx.x;                 // B*S rows
    int b   = row / S_;
    int c   = cond[b];
    int t   = threadIdx.x;

    float x0 = x[(size_t)row * D_ + t];
    float x1 = x[(size_t)row * D_ + t + 256];
    ssum[t] = x0 + x1;
    ssq[t]  = x0 * x0 + x1 * x1;
    __syncthreads();
    for (int o = 128; o > 0; o >>= 1) {
        if (t < o) { ssum[t] += ssum[t + o]; ssq[t] += ssq[t + o]; }
        __syncthreads();
    }
    float mu   = ssum[0] * (1.0f / (float)D_);
    float var  = ssq[0] * (1.0f / (float)D_) - mu * mu;
    float rstd = rsqrtf(var + 1e-5f);

    const float* gg = g  + (size_t)c * D_;
    const float* bb = bt + (size_t)c * D_;
    out[(size_t)row * D_ + t]       = (x0 - mu) * rstd * gg[t]       + bb[t];
    out[(size_t)row * D_ + t + 256] = (x1 - mu) * rstd * gg[t + 256] + bb[t + 256];
}

// ---------------- gumbel softmax epilogue ----------------

__global__ __launch_bounds__(256) void gumbel_kernel(
    const float* __restrict__ logits, float* __restrict__ outg,
    const int* __restrict__ it_raw)
{
    __shared__ float sval[256];
    __shared__ int   sidx[256];
    __shared__ float ssum[256];

    int row = blockIdx.x;
    int t   = threadIdx.x;

    float it;
    { // scalar may be int32 or float32; denorm-range floats indicate int storage
        int i = it_raw[0];
        float f = __int_as_float(i);
        it = (f >= 1e-6f && f <= 1e6f) ? f : (float)i;
    }
    const float* lr = logits + (size_t)row * V_;

    // pass 1: max + argmax of (l+g)*it
    float bm = -3.4e38f; int bi = 0;
    for (int vv = t; vv < V_; vv += 256) {
        float tv = (lr[vv] + gumbel_noise(row, vv)) * it;
        if (tv > bm) { bm = tv; bi = vv; }
    }
    sval[t] = bm; sidx[t] = bi;
    __syncthreads();
    for (int o = 128; o > 0; o >>= 1) {
        if (t < o) {
            if (sval[t + o] > sval[t] ||
                (sval[t + o] == sval[t] && sidx[t + o] < sidx[t])) {
                sval[t] = sval[t + o]; sidx[t] = sidx[t + o];
            }
        }
        __syncthreads();
    }
    float mx = sval[0]; int am = sidx[0];

    // pass 2: denominator
    float acc = 0.0f;
    for (int vv = t; vv < V_; vv += 256)
        acc += expf((lr[vv] + gumbel_noise(row, vv)) * it - mx);
    ssum[t] = acc;
    __syncthreads();
    for (int o = 128; o > 0; o >>= 1) {
        if (t < o) ssum[t] += ssum[t + o];
        __syncthreads();
    }
    float den = ssum[0];

    // pass 3: write (y1 - y) + y
    for (int vv = t; vv < V_; vv += 256) {
        float y  = expf((lr[vv] + gumbel_noise(row, vv)) * it - mx) / den;
        float y1 = (vv == am) ? 1.0f : 0.0f;
        outg[(size_t)row * V_ + vv] = (y1 - y) + y;
    }
}

// ---------------- orchestration ----------------

extern "C" void kernel_launch(void* const* d_in, const int* in_sizes, int n_in,
                              void* d_out, int out_size, void* d_ws, size_t ws_size,
                              hipStream_t stream)
{
    (void)in_sizes; (void)n_in; (void)out_size; (void)ws_size;

    const int*   inputs = (const int*)  d_in[0];
    const int*   cond   = (const int*)  d_in[1];
    const float* mask   = (const float*)d_in[2];
    const int*   itp    = (const int*)  d_in[3];
    const float* emb    = (const float*)d_in[4];
    const float* pos    = (const float*)d_in[5];
    const float* Wq     = (const float*)d_in[6];
    const float* bq     = (const float*)d_in[7];
    const float* Wk     = (const float*)d_in[8];
    const float* bk     = (const float*)d_in[9];
    const float* Wv     = (const float*)d_in[10];
    const float* bv     = (const float*)d_in[11];
    const float* Wo     = (const float*)d_in[12];
    const float* bo     = (const float*)d_in[13];
    const float* W1     = (const float*)d_in[14];
    const float* b1     = (const float*)d_in[15];
    const float* W2     = (const float*)d_in[16];
    const float* b2     = (const float*)d_in[17];
    const float* cn1g   = (const float*)d_in[18];
    const float* cn1b   = (const float*)d_in[19];
    const float* cn2g   = (const float*)d_in[20];
    const float* cn2b   = (const float*)d_in[21];
    const float* ng     = (const float*)d_in[22];
    const float* nb     = (const float*)d_in[23];
    const float* Wout   = (const float*)d_in[24];
    const float* bout   = (const float*)d_in[25];

    float* ws = (float*)d_ws;
    const size_t BSD = (size_t)B_ * S_ * D_;
    float* x  = ws;
    float* q  = x  + BSD;
    float* k  = q  + BSD;
    float* v  = k  + BSD;
    float* a  = v  + BSD;
    float* t2 = a  + BSD;
    float* t1 = t2 + BSD;   // B*S*F

    const int rows = B_ * S_;
    embed_kernel<<<rows, 256, 0, stream>>>(inputs, emb, pos, x);

    dim3 gD(D_ / 64, rows / 64);
    dim3 gF(F_ / 64, rows / 64);
    dim3 gV(V_ / 64, rows / 64);
    const int ropeN = B_ * S_ * H_ * 32;

    for (int l = 0; l < L_; ++l) {
        const float* wq = Wq + (size_t)l * D_ * D_;
        const float* wk = Wk + (size_t)l * D_ * D_;
        const float* wv = Wv + (size_t)l * D_ * D_;
        const float* wo = Wo + (size_t)l * D_ * D_;
        const float* w1 = W1 + (size_t)l * D_ * F_;
        const float* w2 = W2 + (size_t)l * F_ * D_;

        gemm_wmma<0><<<gD, 256, 0, stream>>>(x, wq, bq + (size_t)l * D_, nullptr, q, rows, D_, D_);
        gemm_wmma<0><<<gD, 256, 0, stream>>>(x, wk, bk + (size_t)l * D_, nullptr, k, rows, D_, D_);
        gemm_wmma<0><<<gD, 256, 0, stream>>>(x, wv, bv + (size_t)l * D_, nullptr, v, rows, D_, D_);

        rope_elu_kernel<<<(ropeN + 255) / 256, 256, 0, stream>>>(q, k, mask);
        la_scan_kernel<<<B_ * H_, 64, 0, stream>>>(q, k, v, a);

        gemm_wmma<2><<<gD, 256, 0, stream>>>(a, wo, bo + (size_t)l * D_, x, t2, rows, D_, D_);
        cond_ln_kernel<<<rows, 256, 0, stream>>>(t2, cn1g + (size_t)l * C_ * D_,
                                                 cn1b + (size_t)l * C_ * D_, cond, x);

        gemm_wmma<1><<<gF, 256, 0, stream>>>(x, w1, b1 + (size_t)l * F_, nullptr, t1, rows, F_, D_);
        gemm_wmma<2><<<gD, 256, 0, stream>>>(t1, w2, b2 + (size_t)l * D_, x, t2, rows, D_, F_);
        cond_ln_kernel<<<rows, 256, 0, stream>>>(t2, cn2g + (size_t)l * C_ * D_,
                                                 cn2b + (size_t)l * C_ * D_, cond, x);
    }

    cond_ln_kernel<<<rows, 256, 0, stream>>>(x, ng, nb, cond, t2);

    float* logits = (float*)d_out;
    gemm_wmma<0><<<gV, 256, 0, stream>>>(t2, Wout, bout, nullptr, logits, rows, V_, D_);

    float* outg = logits + (size_t)rows * V_;
    gumbel_kernel<<<rows, 256, 0, stream>>>(logits, outg, itp);
}
